// SimpleDetector_57810259804256
// MI455X (gfx1250) — matrix operand order repaired and checked by reading, not compiled, not run on hardware
//
#include <hip/hip_runtime.h>
#include <hip/hip_bf16.h>

typedef __bf16 bf16_t;
typedef __attribute__((ext_vector_type(4)))  float   v4f;
typedef __attribute__((ext_vector_type(8)))  float   v8f;
typedef __attribute__((ext_vector_type(8)))  __bf16  v8bf;
typedef __attribute__((ext_vector_type(16))) __bf16  v16bf;

#define BATCH   512
#define NBOX    64
#define DIN     2048
#define DOUT    1024
#define KCH     64        // K per chunk = 2 WMMA K-steps
#define LDA     72        // padded LDS row stride (bf16 elems): 144B, 16B-aligned

// ---------------------------------------------------------------------------
// Kernel 1: split W (f32 [2048,1024]) into transposed bf16 hi/lo [1024,2048].
// LDS-tiled 32x32 transpose so both global read and write are coalesced.
// ---------------------------------------------------------------------------
__global__ __launch_bounds__(256) void convert_w_kernel(
    const float* __restrict__ W, bf16_t* __restrict__ Whi, bf16_t* __restrict__ Wlo)
{
    __shared__ float tile[32][33];
    const int k0 = blockIdx.x * 32;
    const int n0 = blockIdx.y * 32;
    const int tr = threadIdx.x >> 5;   // 0..7
    const int tc = threadIdx.x & 31;   // 0..31
    #pragma unroll
    for (int j = 0; j < 4; ++j) {
        const int kr = j * 8 + tr;
        tile[kr][tc] = W[(size_t)(k0 + kr) * DOUT + n0 + tc];
    }
    __syncthreads();
    #pragma unroll
    for (int j = 0; j < 4; ++j) {
        const int nr = j * 8 + tr;
        const float x = tile[tc][nr];
        const bf16_t hi = (bf16_t)x;
        const bf16_t lo = (bf16_t)(x - (float)hi);
        const size_t idx = (size_t)(n0 + nr) * DIN + k0 + tc;
        Whi[idx] = hi;
        Wlo[idx] = lo;
    }
}

// ---------------------------------------------------------------------------
// Kernel 2: per batch (64 rows) fused split-bf16 GEMM + bias + ReLU +
// LayerNorm + masked valid-first repack scatter.
// 512 threads = 16 waves = 4 row-groups(16) x 4 col-groups(256).
// Double-buffered LDS A tile, ONE barrier per 64-K chunk (32 barriers total).
// ---------------------------------------------------------------------------
__global__ __launch_bounds__(512) void fused_detector_kernel(
    const float* __restrict__ X,      // [B*N, 2048] f32
    const int*   __restrict__ mask,   // [B, N]
    const bf16_t* __restrict__ Whi,   // [1024, 2048] bf16 (transposed)
    const bf16_t* __restrict__ Wlo,   // [1024, 2048] bf16 (transposed)
    const float* __restrict__ bias,   // [1024]
    const float* __restrict__ lnw,    // [1024]
    const float* __restrict__ lnb,    // [1024]
    float* __restrict__ out)          // [B, N, 1024]
{
    __shared__ bf16_t sAhi[2][NBOX * LDA];   // double-buffered A tile, hi part
    __shared__ bf16_t sAlo[2][NBOX * LDA];   // lo part
    __shared__ float  sRedS[NBOX * 4];       // per-row per-colgroup partial sum
    __shared__ float  sRedQ[NBOX * 4];       // per-row per-colgroup partial sumsq
    __shared__ int    sMask[NBOX];

    const int tid  = threadIdx.x;
    const int lane = tid & 31;
    const int wave = tid >> 5;         // 0..15
    const int rg   = wave >> 2;        // row group 0..3  -> rows rg*16..+15
    const int cg   = wave & 3;         // col group 0..3  -> cols cg*256..+255
    const int g    = lane >> 4;        // lane half
    const int lIdx = lane & 15;

    const int bb = blockIdx.x;                 // one batch per block
    const size_t rowBase = (size_t)bb * NBOX;

    if (tid < NBOX) sMask[tid] = mask[bb * NBOX + tid];

    v8f acc[16] = {};

    // cooperative staging: 512 threads cover 64 rows x 64 K, 8 f32 each
    const int ldRow = tid >> 3;          // 0..63
    const int ldK   = (tid & 7) << 3;    // 0,8,...,56
    const float* Xrow = X + (rowBase + ldRow) * DIN;

    const int colBase = cg * 256;
    const int mRow    = rg * 16 + lIdx;  // A-matrix M row for this lane

    v4f x0 = *(const v4f*)(Xrow + ldK);      // prefetch first chunk
    v4f x1 = *(const v4f*)(Xrow + ldK + 4);

    int q = 0;
    for (int kc = 0; kc < DIN; kc += KCH) {
        // split f32 -> bf16 hi/lo (8 elems) and store 16B each to buffer q
        v8bf hi8, lo8;
        #pragma unroll
        for (int e = 0; e < 4; ++e) {
            float xa = x0[e], xb = x1[e];
            bf16_t ha = (bf16_t)xa, hb = (bf16_t)xb;
            hi8[e]     = ha;
            hi8[e + 4] = hb;
            lo8[e]     = (bf16_t)(xa - (float)ha);
            lo8[e + 4] = (bf16_t)(xb - (float)hb);
        }
        *(v8bf*)(sAhi[q] + ldRow * LDA + ldK) = hi8;
        *(v8bf*)(sAlo[q] + ldRow * LDA + ldK) = lo8;
        __syncthreads();   // single barrier: buffer q now readable; q^1 free

        // issue next chunk's X load early to hide latency behind WMMA burst
        if (kc + KCH < DIN) {
            x0 = *(const v4f*)(Xrow + kc + KCH + ldK);
            x1 = *(const v4f*)(Xrow + kc + KCH + ldK + 4);
        }

        #pragma unroll
        for (int ks = 0; ks < 2; ++ks) {
            const int kb = ks * 32;
            // A fragments: lane m=mRow; VGPR0-3 = K[8g..+7], VGPR4-7 = K[16+8g..+7]
            union { v16bf v; v8bf h[2]; } ahi, alo;
            const bf16_t* aH = sAhi[q] + mRow * LDA + kb;
            const bf16_t* aL = sAlo[q] + mRow * LDA + kb;
            ahi.h[0] = *(const v8bf*)(aH + 8 * g);
            ahi.h[1] = *(const v8bf*)(aH + 16 + 8 * g);
            alo.h[0] = *(const v8bf*)(aL + 8 * g);
            alo.h[1] = *(const v8bf*)(aL + 16 + 8 * g);

            #pragma unroll
            for (int t = 0; t < 16; ++t) {
                const int n = colBase + t * 16 + lIdx;
                // B fragment: lane n; VGPR0-3 = K[8g..+7], VGPR4-7 = K[16+8g..+7]
                const bf16_t* wh = Whi + (size_t)n * DIN + kc + kb;
                const bf16_t* wl = Wlo + (size_t)n * DIN + kc + kb;
                union { v16bf v; v8bf h[2]; } bhi, blo;
                bhi.h[0] = *(const v8bf*)(wh + 8 * g);
                bhi.h[1] = *(const v8bf*)(wh + 16 + 8 * g);
                blo.h[0] = *(const v8bf*)(wl + 8 * g);
                blo.h[1] = *(const v8bf*)(wl + 16 + 8 * g);
                // split product: hi*hi + hi*lo + lo*hi  (fp32 accumulate)
                acc[t] = __builtin_amdgcn_wmma_f32_16x16x32_bf16(
                    false, ahi.v, false, bhi.v, (short)0, acc[t], false, false);
                acc[t] = __builtin_amdgcn_wmma_f32_16x16x32_bf16(
                    false, ahi.v, false, blo.v, (short)0, acc[t], false, false);
                acc[t] = __builtin_amdgcn_wmma_f32_16x16x32_bf16(
                    false, alo.v, false, bhi.v, (short)0, acc[t], false, false);
            }
        }
        q ^= 1;
    }

    // ---- epilogue: bias + ReLU, row stats via shuffles + tiny LDS reduce ----
    // acc[t][v] = h(row rg*16 + v + 8g, col colBase + t*16 + lIdx)
    float biasv[16];
    float sumv[8] = {}, sqv[8] = {};
    #pragma unroll
    for (int t = 0; t < 16; ++t) {
        const int col = colBase + t * 16 + lIdx;
        biasv[t] = bias[col];
        #pragma unroll
        for (int v = 0; v < 8; ++v) {
            const float h = fmaxf(acc[t][v] + biasv[t], 0.0f);
            sumv[v] += h;
            sqv[v]  += h * h;
        }
    }
    #pragma unroll
    for (int v = 0; v < 8; ++v) {
        #pragma unroll
        for (int off = 8; off > 0; off >>= 1) {
            sumv[v] += __shfl_xor(sumv[v], off, 16);
            sqv[v]  += __shfl_xor(sqv[v],  off, 16);
        }
    }
    if (lIdx == 0) {
        #pragma unroll
        for (int v = 0; v < 8; ++v) {
            const int m = rg * 16 + v + 8 * g;
            sRedS[m * 4 + cg] = sumv[v];
            sRedQ[m * 4 + cg] = sqv[v];
        }
    }
    __syncthreads();

    int nvt = 0;
    #pragma unroll 8
    for (int j = 0; j < NBOX; ++j) nvt += sMask[j];

    float meanv[8], rstdv[8];
    int   destv[8], mv[8];
    #pragma unroll
    for (int v = 0; v < 8; ++v) {
        const int m = rg * 16 + v + 8 * g;     // row in batch
        const float s = sRedS[m*4+0] + sRedS[m*4+1] + sRedS[m*4+2] + sRedS[m*4+3];
        const float qq = sRedQ[m*4+0] + sRedQ[m*4+1] + sRedQ[m*4+2] + sRedQ[m*4+3];
        const float mean = s * (1.0f / 1024.0f);
        const float var  = qq * (1.0f / 1024.0f) - mean * mean;
        meanv[v] = mean;
        rstdv[v] = __frsqrt_rn(var + 1e-12f);

        int nvb = 0;
        for (int j = 0; j < m; ++j) nvb += sMask[j];
        mv[v]    = sMask[m];
        // valid rows compact to front (stable); invalid rows fill zeroed tail
        destv[v] = mv[v] ? nvb : (nvt + (m - nvb));
    }

    // ---- normalize + scatter-store (16 lanes -> 64B contiguous per store) ----
    #pragma unroll
    for (int t = 0; t < 16; ++t) {
        const int col = colBase + t * 16 + lIdx;
        const float lw = lnw[col];
        const float lb = lnb[col];
        #pragma unroll
        for (int v = 0; v < 8; ++v) {
            float h = fmaxf(acc[t][v] + biasv[t], 0.0f);
            const float val = mv[v] ? ((h - meanv[v]) * rstdv[v] * lw + lb) : 0.0f;
            out[(rowBase + destv[v]) * DOUT + col] = val;
        }
    }
}

extern "C" void kernel_launch(void* const* d_in, const int* in_sizes, int n_in,
                              void* d_out, int out_size, void* d_ws, size_t ws_size,
                              hipStream_t stream) {
    // setup_inputs order: img_feats, boxes, box_mask, obj_labels, W, b, ln_w, ln_b
    const float* X    = (const float*)d_in[0];
    const int*   mask = (const int*)d_in[2];
    const float* W    = (const float*)d_in[4];
    const float* bias = (const float*)d_in[5];
    const float* lnw  = (const float*)d_in[6];
    const float* lnb  = (const float*)d_in[7];
    float* out = (float*)d_out;

    // workspace: Wt_hi [1024*2048] bf16 (4MB) + Wt_lo (4MB)
    bf16_t* Whi = (bf16_t*)d_ws;
    bf16_t* Wlo = Whi + (size_t)DOUT * DIN;

    convert_w_kernel<<<dim3(DIN / 32, DOUT / 32), dim3(256), 0, stream>>>(W, Whi, Wlo);
    fused_detector_kernel<<<dim3(BATCH), dim3(512), 0, stream>>>(
        X, mask, Whi, Wlo, bias, lnw, lnb, out);
}
